// pyg_GATLayer_2010044694728
// MI455X (gfx1250) — matrix operand order repaired and checked by reading, not compiled, not run on hardware
//
#include <hip/hip_runtime.h>
#include <hip/hip_bf16.h>

typedef __bf16 bf16_t;
typedef bf16_t v16bf __attribute__((ext_vector_type(16)));
typedef bf16_t v8bf  __attribute__((ext_vector_type(8)));
typedef bf16_t v4bf  __attribute__((ext_vector_type(4)));
typedef float  v8f   __attribute__((ext_vector_type(8)));
typedef int    v4i   __attribute__((ext_vector_type(4)));

#define AS1 __attribute__((address_space(1)))
#define AS3 __attribute__((address_space(3)))

#define N_NODES 100000
#define N_EDGES 1600000
#define HC 128           // H*C = 4*32
#define NEG_SLOPE 0.2f
#define MT 64            // GEMM row tile per block
#define KP 136           // padded LDS stride in bf16 elems (272B -> 4-bank rotation)

#define HAVE_ASYNC_LDS __has_builtin(__builtin_amdgcn_global_load_async_to_lds_b128)

__device__ __forceinline__ float lrelu(float x) { return x > 0.f ? x : NEG_SLOPE * x; }

__device__ __forceinline__ void atomicMaxF(float* addr, float v) {
    if (v >= 0.f) atomicMax((int*)addr, __float_as_int(v));
    else          atomicMin((unsigned int*)addr, __float_as_uint(v));
}

__device__ __forceinline__ void wait_async_zero() {
#if __has_builtin(__builtin_amdgcn_s_wait_asynccnt)
    __builtin_amdgcn_s_wait_asynccnt(0);
#else
    asm volatile("s_wait_asynccnt 0x0" ::: "memory");
#endif
}

// ---------------- weight fp32 -> bf16 pre-convert ----------------
__global__ void k_convert_w(const float* __restrict__ wsrc, const float* __restrict__ wres,
                            bf16_t* __restrict__ wsrc_b, bf16_t* __restrict__ wres_b) {
    int i = blockIdx.x * blockDim.x + threadIdx.x;
    if (i < HC * HC) {
        wsrc_b[i] = (bf16_t)wsrc[i];
        wres_b[i] = (bf16_t)wres[i];
    }
}

// ---------------- feats fp32 -> bf16 pre-convert ----------------
__global__ void k_convert_x(const float* __restrict__ feats, bf16_t* __restrict__ featsb) {
    size_t i = ((size_t)blockIdx.x * blockDim.x + threadIdx.x) * 4;
    size_t stride = (size_t)gridDim.x * blockDim.x * 4;
    const size_t tot = (size_t)N_NODES * HC;   // multiple of 4
    for (; i < tot; i += stride) {
        float4 v = *(const float4*)(feats + i);
        v4bf o;
        o[0] = (bf16_t)v.x; o[1] = (bf16_t)v.y; o[2] = (bf16_t)v.z; o[3] = (bf16_t)v.w;
        *(v4bf*)(featsb + i) = o;
    }
}

// ---------------- per-call scratch init (deterministic) ----------------
__global__ void k_init(float* __restrict__ m, float* __restrict__ s,
                       float* __restrict__ chsum, float* __restrict__ chsq) {
    int i = blockIdx.x * blockDim.x + threadIdx.x;
    int stride = gridDim.x * blockDim.x;
    for (int j = i; j < N_NODES * 4; j += stride) { m[j] = -3.0e38f; s[j] = 0.f; }
    if (i < HC) { chsum[i] = 0.f; chsq[i] = 0.f; }
}

// ---------------- dual GEMM: xp = feats@Wsrc^T (bf16 out), res = feats@Wres^T (f32 out) ----------------
__global__ __launch_bounds__(256) void k_gemm(const bf16_t* __restrict__ featsb,
                                              const bf16_t* __restrict__ wsrc_b,
                                              const bf16_t* __restrict__ wres_b,
                                              bf16_t* __restrict__ xp,
                                              float* __restrict__ resbuf) {
    __shared__ bf16_t sA[MT * KP];   // 64 x 128 activations (bf16, padded)
    __shared__ bf16_t sW[HC * KP];   // 128 x 128 weight (bf16, padded)
    const int t = threadIdx.x;
    const int mat = blockIdx.y;
    const bf16_t* __restrict__ wb = (mat == 0) ? wsrc_b : wres_b;
    const int row0 = blockIdx.x * MT;

#if HAVE_ASYNC_LDS
    // A: 64 rows x 16 chunks (16B) = 1024 async copies, DMA directly into LDS
#pragma unroll
    for (int it = 0; it < 4; ++it) {
        int q = t + it * 256;
        int r = q >> 4;              // 16 chunks per 128-bf16 row
        int c = (q & 15) << 3;
        int gr = row0 + r;
        if (gr < N_NODES) {
            __builtin_amdgcn_global_load_async_to_lds_b128(
                (AS1 v4i*)(featsb + (size_t)gr * HC + c),
                (AS3 v4i*)&sA[r * KP + c], 0, 0);
        }
    }
    // W: 128 rows x 16 chunks = 2048 async copies
#pragma unroll
    for (int it = 0; it < 8; ++it) {
        int q = t + it * 256;
        int r = q >> 4;
        int c = (q & 15) << 3;
        __builtin_amdgcn_global_load_async_to_lds_b128(
            (AS1 v4i*)(wb + (size_t)r * HC + c),
            (AS3 v4i*)&sW[r * KP + c], 0, 0);
    }
    wait_async_zero();
#else
    // fallback: stage via VGPRs
#pragma unroll
    for (int it = 0; it < 4; ++it) {
        int q = t + it * 256;
        int r = q >> 4;
        int c = (q & 15) << 3;
        int gr = row0 + r;
        if (gr < N_NODES)
            *(v8bf*)&sA[r * KP + c] = *(const v8bf*)(featsb + (size_t)gr * HC + c);
    }
#pragma unroll
    for (int it = 0; it < 8; ++it) {
        int q = t + it * 256;
        int r = q >> 4;
        int c = (q & 15) << 3;
        *(v8bf*)&sW[r * KP + c] = *(const v8bf*)(wb + (size_t)r * HC + c);
    }
#endif
    __syncthreads();

    const int wv = t >> 5;
    const int lane = t & 31;
    const int strip = wv >> 1;          // 4 strips x 16 rows
    const int chalf = (wv & 1) * 4;     // 4 col-tiles per wave
    const int g = lane >> 4;            // half-wave group
    const int l15 = lane & 15;
    const int arow = strip * 16 + l15;

    v8f acc[4] = {};
#pragma unroll
    for (int ks = 0; ks < 4; ++ks) {
        int ka = ks * 32 + g * 8;
        v8bf alo = *(const v8bf*)&sA[arow * KP + ka];
        v8bf ahi = *(const v8bf*)&sA[arow * KP + ka + 16];
        v16bf afrag = __builtin_shufflevector(alo, ahi, 0,1,2,3,4,5,6,7,8,9,10,11,12,13,14,15);
        int kb = ks * 32 + g * 8;
#pragma unroll
        for (int ct = 0; ct < 4; ++ct) {
            int n = (chalf + ct) * 16 + l15;
            v8bf blo = *(const v8bf*)&sW[n * KP + kb];
            v8bf bhi = *(const v8bf*)&sW[n * KP + kb + 16];
            v16bf bfrag = __builtin_shufflevector(blo, bhi, 0,1,2,3,4,5,6,7,8,9,10,11,12,13,14,15);
            acc[ct] = __builtin_amdgcn_wmma_f32_16x16x32_bf16(
                false, afrag, false, bfrag, (short)0, acc[ct], false, false);
        }
    }

#pragma unroll
    for (int ct = 0; ct < 4; ++ct) {
        int ncol = (chalf + ct) * 16 + l15;
#pragma unroll
        for (int i = 0; i < 8; ++i) {
            int gr = row0 + strip * 16 + g * 8 + i;
            if (gr < N_NODES) {
                if (mat == 0) xp[(size_t)gr * HC + ncol] = (bf16_t)acc[ct][i];
                else          resbuf[(size_t)gr * HC + ncol] = acc[ct][i];
            }
        }
    }
}

// ---------------- attention coefficients a_src/a_dst ----------------
__global__ void k_att(const bf16_t* __restrict__ xp, const float* __restrict__ att_s,
                      const float* __restrict__ att_d, float* __restrict__ a_s,
                      float* __restrict__ a_d) {
    int i = blockIdx.x * blockDim.x + threadIdx.x;
    if (i >= N_NODES * 4) return;
    int n = i >> 2, h = i & 3;
    const bf16_t* x = xp + (size_t)n * HC + h * 32;
    const float* ws = att_s + h * 32;
    const float* wd = att_d + h * 32;
    float ss = 0.f, sd = 0.f;
#pragma unroll
    for (int c = 0; c < 32; ++c) {
        float xv = (float)x[c];
        ss += xv * ws[c]; sd += xv * wd[c];
    }
    a_s[i] = ss; a_d[i] = sd;
}

// ---------------- edge pass 1: segment max ----------------
__global__ void k_edge_max(const int* __restrict__ ei, const float* __restrict__ a_s,
                           const float* __restrict__ a_d, float* __restrict__ m) {
    int i = blockIdx.x * blockDim.x + threadIdx.x;
    int stride = gridDim.x * blockDim.x;
    const int tot = N_EDGES + N_NODES;
    for (; i < tot; i += stride) {
        int s, d;
        if (i < N_EDGES) { s = ei[i]; d = ei[N_EDGES + i]; } else { s = d = i - N_EDGES; }
        float4 as = ((const float4*)a_s)[s];
        float4 ad = ((const float4*)a_d)[d];
        atomicMaxF(&m[d * 4 + 0], lrelu(as.x + ad.x));
        atomicMaxF(&m[d * 4 + 1], lrelu(as.y + ad.y));
        atomicMaxF(&m[d * 4 + 2], lrelu(as.z + ad.z));
        atomicMaxF(&m[d * 4 + 3], lrelu(as.w + ad.w));
    }
}

// ---------------- edge pass 2: segment sum of exp ----------------
__global__ void k_edge_sum(const int* __restrict__ ei, const float* __restrict__ a_s,
                           const float* __restrict__ a_d, const float* __restrict__ m,
                           float* __restrict__ ssum) {
    int i = blockIdx.x * blockDim.x + threadIdx.x;
    int stride = gridDim.x * blockDim.x;
    const int tot = N_EDGES + N_NODES;
    for (; i < tot; i += stride) {
        int s, d;
        if (i < N_EDGES) { s = ei[i]; d = ei[N_EDGES + i]; } else { s = d = i - N_EDGES; }
        float4 as = ((const float4*)a_s)[s];
        float4 ad = ((const float4*)a_d)[d];
        float4 mv = ((const float4*)m)[d];
        atomicAdd(&ssum[d * 4 + 0], __expf(lrelu(as.x + ad.x) - mv.x));
        atomicAdd(&ssum[d * 4 + 1], __expf(lrelu(as.y + ad.y) - mv.y));
        atomicAdd(&ssum[d * 4 + 2], __expf(lrelu(as.z + ad.z) - mv.z));
        atomicAdd(&ssum[d * 4 + 3], __expf(lrelu(as.w + ad.w) - mv.w));
    }
}

// ---------------- edge pass 3: weighted scatter-add (1 wave / edge, 4 ch / lane) ----------------
__global__ __launch_bounds__(256) void k_edge_agg(const int* __restrict__ ei,
                                                  const float* __restrict__ a_s,
                                                  const float* __restrict__ a_d,
                                                  const float* __restrict__ m,
                                                  const float* __restrict__ ssum,
                                                  const bf16_t* __restrict__ xp,
                                                  float* __restrict__ resbuf) {
    int wid = (blockIdx.x * blockDim.x + threadIdx.x) >> 5;
    int lane = threadIdx.x & 31;
    int nw = (gridDim.x * blockDim.x) >> 5;
    const int tot = N_EDGES + N_NODES;
    const int h = lane >> 3;        // 8 lanes per head (32 ch / head, 4 ch / lane)
    const int c = lane << 2;
    for (int e = wid; e < tot; e += nw) {
        int s, d;
        if (e < N_EDGES) { s = ei[e]; d = ei[N_EDGES + e]; } else { s = d = e - N_EDGES; }
        float ev = lrelu(a_s[s * 4 + h] + a_d[d * 4 + h]);
        float alpha = __expf(ev - m[d * 4 + h]) / (ssum[d * 4 + h] + 1e-16f);
        v4bf xv = *(const v4bf*)(xp + (size_t)s * HC + c);
        float* orow = resbuf + (size_t)d * HC + c;
        atomicAdd(orow + 0, alpha * (float)xv[0]);
        atomicAdd(orow + 1, alpha * (float)xv[1]);
        atomicAdd(orow + 2, alpha * (float)xv[2]);
        atomicAdd(orow + 3, alpha * (float)xv[3]);
    }
}

// ---------------- batch-norm stats (per-channel sum / sumsq) ----------------
__global__ __launch_bounds__(256) void k_stats(const float* __restrict__ resbuf,
                                               const float* __restrict__ bias,
                                               float* __restrict__ chsum,
                                               float* __restrict__ chsq) {
    __shared__ float lsum[HC], lsq[HC];
    int t = threadIdx.x;
    if (t < HC) { lsum[t] = 0.f; lsq[t] = 0.f; }
    __syncthreads();
    size_t stride = (size_t)gridDim.x * blockDim.x;      // multiple of 128
    size_t i = (size_t)blockIdx.x * blockDim.x + t;
    int c = (int)(i & (HC - 1));
    float b = bias[c];
    float ps = 0.f, pq = 0.f;
    const size_t tot = (size_t)N_NODES * HC;
    for (; i < tot; i += stride) {
        float v = resbuf[i] + b;
        ps += v; pq += v * v;
    }
    atomicAdd(&lsum[c], ps);
    atomicAdd(&lsq[c], pq);
    __syncthreads();
    if (t < HC) { atomicAdd(&chsum[t], lsum[t]); atomicAdd(&chsq[t], lsq[t]); }
}

__global__ void k_stats_final(const float* __restrict__ chsum, const float* __restrict__ chsq,
                              float* __restrict__ mu, float* __restrict__ inv) {
    int t = threadIdx.x;
    if (t < HC) {
        float m = chsum[t] / (float)N_NODES;
        float var = chsq[t] / (float)N_NODES - m * m;
        mu[t] = m;
        inv[t] = rsqrtf(var + 1e-5f);
    }
}

// ---------------- normalize + affine -> d_out ----------------
__global__ void k_norm(const float* __restrict__ resbuf, const float* __restrict__ bias,
                       const float* __restrict__ mu, const float* __restrict__ inv,
                       const float* __restrict__ gma, const float* __restrict__ bta,
                       float* __restrict__ out) {
    size_t i = (size_t)blockIdx.x * blockDim.x + threadIdx.x;
    size_t stride = (size_t)gridDim.x * blockDim.x;
    const size_t tot = (size_t)N_NODES * HC;
    for (; i < tot; i += stride) {
        int c = (int)(i & (HC - 1));
        out[i] = (resbuf[i] + bias[c] - mu[c]) * inv[c] * gma[c] + bta[c];
    }
}

extern "C" void kernel_launch(void* const* d_in, const int* in_sizes, int n_in,
                              void* d_out, int out_size, void* d_ws, size_t ws_size,
                              hipStream_t stream) {
    const float* feats   = (const float*)d_in[0];
    const int*   ei      = (const int*)d_in[1];     // [2, E] int32
    const float* Wsrc    = (const float*)d_in[2];
    const float* att_src = (const float*)d_in[3];
    const float* att_dst = (const float*)d_in[4];
    const float* Wres    = (const float*)d_in[5];
    const float* bias    = (const float*)d_in[6];
    const float* gma     = (const float*)d_in[7];
    const float* bta     = (const float*)d_in[8];
    float* out = (float*)d_out;

    char* ws = (char*)d_ws;
    size_t off = 0;
    auto carve = [&](size_t bytes) -> void* {
        void* p = ws + off;
        off += (bytes + 255) & ~(size_t)255;
        return p;
    };
    bf16_t* featsb = (bf16_t*)carve((size_t)N_NODES * HC * sizeof(bf16_t));
    bf16_t* xp     = (bf16_t*)carve((size_t)N_NODES * HC * sizeof(bf16_t));
    float*  resbuf = (float*) carve((size_t)N_NODES * HC * sizeof(float));
    float*  a_s    = (float*) carve((size_t)N_NODES * 4 * sizeof(float));
    float*  a_d    = (float*) carve((size_t)N_NODES * 4 * sizeof(float));
    float*  mbuf   = (float*) carve((size_t)N_NODES * 4 * sizeof(float));
    float*  sbuf   = (float*) carve((size_t)N_NODES * 4 * sizeof(float));
    bf16_t* wsrc_b = (bf16_t*)carve((size_t)HC * HC * sizeof(bf16_t));
    bf16_t* wres_b = (bf16_t*)carve((size_t)HC * HC * sizeof(bf16_t));
    float*  chsum  = (float*) carve(HC * sizeof(float));
    float*  chsq   = (float*) carve(HC * sizeof(float));
    float*  mu     = (float*) carve(HC * sizeof(float));
    float*  inv    = (float*) carve(HC * sizeof(float));

    k_convert_w<<<(HC * HC + 255) / 256, 256, 0, stream>>>(Wsrc, Wres, wsrc_b, wres_b);
    k_convert_x<<<4096, 256, 0, stream>>>(feats, featsb);
    k_init<<<1563, 256, 0, stream>>>(mbuf, sbuf, chsum, chsq);
    k_gemm<<<dim3((N_NODES + MT - 1) / MT, 2), 256, 0, stream>>>(featsb, wsrc_b, wres_b, xp, resbuf);
    k_att<<<(N_NODES * 4 + 255) / 256, 256, 0, stream>>>(xp, att_src, att_dst, a_s, a_d);
    k_edge_max<<<4096, 256, 0, stream>>>(ei, a_s, a_d, mbuf);
    k_edge_sum<<<4096, 256, 0, stream>>>(ei, a_s, a_d, mbuf, sbuf);
    k_edge_agg<<<8192, 256, 0, stream>>>(ei, a_s, a_d, mbuf, sbuf, xp, resbuf);
    k_stats<<<1024, 256, 0, stream>>>(resbuf, bias, chsum, chsq);
    k_stats_final<<<1, 128, 0, stream>>>(chsum, chsq, mu, inv);
    k_norm<<<4096, 256, 0, stream>>>(resbuf, bias, mu, inv, gma, bta, out);
}
